// KoLeoLossDistributed_6622839570986
// MI455X (gfx1250) — matrix operand order repaired and checked by reading, not compiled, not run on hardware
//
#include <hip/hip_runtime.h>
#include <hip/hip_bf16.h>
#include <math.h>

// ---------------------------------------------------------------------------
// KoLeo loss, B=8192, D=256, TOPK=1, single device.
//   m_i = max_{j!=i} <xn_i, xn_j>   (xn = row-normalized input)
//   loss = -mean( log( sqrt(max(2-2*m_i,0)) + 2*EPS ) )
// Pipeline:
//   K1: normalize rows (f32) -> bf16 matrix in ws, init row-max keys
//   K2: fused bf16 WMMA Gram + row/col max, LOWER-TRIANGULAR blocks only
//       (symmetry: dots[i][j]==dots[j][i] -> 1.97x fewer FLOPs)
//   K3: decode keys -> distances -> -log -> mean
// ---------------------------------------------------------------------------

typedef __attribute__((ext_vector_type(16))) __bf16 bf16x16;
typedef __attribute__((ext_vector_type(8)))  __bf16 bf16x8;
typedef __attribute__((ext_vector_type(4)))  __bf16 bf16x4;
typedef __attribute__((ext_vector_type(8)))  float  f32x8;

#define BB 8192
#define DD 256
#define EPSV 1e-8f
#define NTB 64                      // 8192/128 tile-blocks per dim
#define NTRI (NTB * (NTB + 1) / 2)  // 2080 lower-triangular blocks

__device__ __forceinline__ unsigned encodeKey(float f) {
    unsigned u = __float_as_uint(f);
    return (u >> 31) ? ~u : (u | 0x80000000u);   // order-preserving float->uint
}

// ---------------- K1: row L2-normalize + f32->bf16, one wave per row -------
__global__ __launch_bounds__(256)
void koleo_normalize(const float* __restrict__ S,
                     __bf16* __restrict__ Xb,
                     unsigned* __restrict__ keys) {
    const int wave = threadIdx.x >> 5;
    const int lane = threadIdx.x & 31;
    const int row  = blockIdx.x * 8 + wave;

    const float4* src = (const float4*)(S + (size_t)row * DD);
    float4 p0 = src[lane];        // elements [lane*4 .. lane*4+3]
    float4 p1 = src[lane + 32];   // elements [128 + lane*4 ..]

    float s = p0.x*p0.x + p0.y*p0.y + p0.z*p0.z + p0.w*p0.w
            + p1.x*p1.x + p1.y*p1.y + p1.z*p1.z + p1.w*p1.w;
    #pragma unroll
    for (int m = 16; m >= 1; m >>= 1) s += __shfl_xor(s, m, 32);

    const float inv = 1.0f / (sqrtf(s) + EPSV);

    bf16x4 o0 = { (__bf16)(p0.x*inv), (__bf16)(p0.y*inv),
                  (__bf16)(p0.z*inv), (__bf16)(p0.w*inv) };
    bf16x4 o1 = { (__bf16)(p1.x*inv), (__bf16)(p1.y*inv),
                  (__bf16)(p1.z*inv), (__bf16)(p1.w*inv) };
    bf16x4* dst = (bf16x4*)(Xb + (size_t)row * DD);
    dst[lane]      = o0;
    dst[lane + 32] = o1;

    if (lane == 0) keys[row] = 0u;   // < encode(-1.0): safe lower bound
}

// ---------------- K2: fused Gram GEMM + row/col max (triangular) -----------
// 128x128 C tile / block, 8 waves (4 along M, 2 along N).
// Each wave: 2 (M) x 4 (N) tiles of v_wmma_f32_16x16x32_bf16.
// Off-diagonal blocks feed BOTH keys[row] (max over N) and keys[col]
// (max over M); diagonal blocks mask self-pairs and use row-max only.
__global__ __launch_bounds__(256)
void koleo_gram_rowmax(const __bf16* __restrict__ Xb,
                       unsigned* __restrict__ keys) {
    // Decode 1D block id -> lower-triangular (bi >= bj) tile coords.
    const int t  = blockIdx.x;
    int bi = (int)((sqrtf(8.0f * (float)t + 1.0f) - 1.0f) * 0.5f);
    while ((bi + 1) * (bi + 2) / 2 <= t) ++bi;
    while (bi * (bi + 1) / 2 > t)       --bi;
    const int bj   = t - bi * (bi + 1) / 2;
    const int row0 = bi * 128;
    const int col0 = bj * 128;
    const bool diag = (bi == bj);

    const int lane = threadIdx.x & 31;
    const int wid  = threadIdx.x >> 5;
    const int wm   = wid & 3;           // 0..3 : 32-row slab
    const int wn   = wid >> 2;          // 0..1 : 64-col slab

    const int lmod = lane & 15;
    const int lhi  = lane >> 4;         // 0 or 1 (lane half)

    // A fragment rows (M = lane%16 within 16-row subtile), K split per half.
    const __bf16* arow[2];
    #pragma unroll
    for (int i = 0; i < 2; ++i)
        arow[i] = Xb + (size_t)(row0 + wm*32 + i*16 + lmod) * DD;
    // B fragment "columns" are rows of X (Gram symmetry): contiguous K loads.
    const __bf16* brow[4];
    #pragma unroll
    for (int j = 0; j < 4; ++j)
        brow[j] = Xb + (size_t)(col0 + wn*64 + j*16 + lmod) * DD;

    const int akb = lhi * 8;    // A: K offset 0 or 8 inside each 16-block
    const int bkb = lhi * 8;    // B: K offset 0 or 8 inside each 16-block (matches A)

    f32x8 acc[2][4] = {};

    union Frag { bf16x16 v; bf16x8 h[2]; };

    #pragma unroll
    for (int k0 = 0; k0 < DD; k0 += 32) {
        Frag a[2], b[4];
        #pragma unroll
        for (int i = 0; i < 2; ++i) {
            a[i].h[0] = *(const bf16x8*)(arow[i] + k0 + akb);
            a[i].h[1] = *(const bf16x8*)(arow[i] + k0 + 16 + akb);
        }
        #pragma unroll
        for (int j = 0; j < 4; ++j) {
            b[j].h[0] = *(const bf16x8*)(brow[j] + k0 + bkb);
            b[j].h[1] = *(const bf16x8*)(brow[j] + k0 + 16 + bkb);
        }
        #pragma unroll
        for (int i = 0; i < 2; ++i)
            #pragma unroll
            for (int j = 0; j < 4; ++j)
                acc[i][j] = __builtin_amdgcn_wmma_f32_16x16x32_bf16(
                    false, a[i].v, false, b[j].v,
                    (short)0, acc[i][j], false, false);
    }

    // ---- Row-max epilogue (diagonal mask only matters when bi==bj).
    // C layout: VGPR r, lane L -> M = r + 8*(L>>4), N = L&15 (within 16x16).
    #pragma unroll
    for (int i = 0; i < 2; ++i) {
        const int gmb = row0 + wm*32 + i*16 + 8*lhi;
        #pragma unroll
        for (int r = 0; r < 8; ++r) {
            const int gm = gmb + r;
            float v = -2.0f;
            #pragma unroll
            for (int j = 0; j < 4; ++j) {
                const int gn = col0 + wn*64 + j*16 + lmod;
                float c = acc[i][j][r];
                if (gm == gn) c = -2.0f;          // mask self-pair
                v = fmaxf(v, c);
            }
            #pragma unroll
            for (int m = 1; m < 16; m <<= 1)
                v = fmaxf(v, __shfl_xor(v, m, 32));
            if (lmod == 0)
                atomicMax(&keys[gm], encodeKey(v));
        }
    }

    // ---- Column-max epilogue: mirrors the transposed tile into keys[col].
    // Skip on diagonal blocks (row-max already saw the full square there).
    if (!diag) {
        #pragma unroll
        for (int j = 0; j < 4; ++j) {
            float cm = -2.0f;
            #pragma unroll
            for (int i = 0; i < 2; ++i)
                #pragma unroll
                for (int r = 0; r < 8; ++r)
                    cm = fmaxf(cm, acc[i][j][r]);
            cm = fmaxf(cm, __shfl_xor(cm, 16, 32));   // merge M lane-halves
            if (lhi == 0) {
                const int gn = col0 + wn*64 + j*16 + lmod;
                atomicMax(&keys[gn], encodeKey(cm));
            }
        }
    }
}

// ---------------- K3: loss reduction ---------------------------------------
__global__ __launch_bounds__(256)
void koleo_loss(const unsigned* __restrict__ keys, float* __restrict__ out) {
    __shared__ float red[256];
    float sum = 0.0f;
    for (int r = threadIdx.x; r < BB; r += 256) {
        unsigned u = keys[r];
        u = (u & 0x80000000u) ? (u ^ 0x80000000u) : ~u;   // decode
        const float m = __uint_as_float(u);
        const float dist = sqrtf(fmaxf(2.0f - 2.0f * m, 0.0f)) + EPSV;
        sum += -logf(dist + EPSV);
    }
    red[threadIdx.x] = sum;
    __syncthreads();
    for (int s = 128; s > 0; s >>= 1) {
        if (threadIdx.x < s) red[threadIdx.x] += red[threadIdx.x + s];
        __syncthreads();
    }
    if (threadIdx.x == 0) out[0] = red[0] / (float)BB;
}

// ---------------------------------------------------------------------------
extern "C" void kernel_launch(void* const* d_in, const int* in_sizes, int n_in,
                              void* d_out, int out_size, void* d_ws, size_t ws_size,
                              hipStream_t stream) {
    (void)in_sizes; (void)n_in; (void)out_size; (void)ws_size;

    const float* S = (const float*)d_in[0];            // [8192, 256] f32
    __bf16*   Xb   = (__bf16*)d_ws;                    // 4 MB bf16 normalized X
    unsigned* keys = (unsigned*)((char*)d_ws + (size_t)BB * DD * sizeof(__bf16));
    float*    out  = (float*)d_out;

    koleo_normalize<<<BB / 8, 256, 0, stream>>>(S, Xb, keys);
    koleo_gram_rowmax<<<NTRI, 256, 0, stream>>>(Xb, keys);
    koleo_loss<<<1, 256, 0, stream>>>(keys, out);
}
